// GNNModel_7378753815013
// MI455X (gfx1250) — compile-verified, matrix operand order checked by reading, not run on hardware
//
#include <hip/hip_runtime.h>
#include <hip/hip_bf16.h>
#include <math.h>

typedef __attribute__((ext_vector_type(16))) __bf16 v16bf;
typedef __attribute__((ext_vector_type(8)))  float  v8f;

#define NNODE 30000
#define NEDGE 120000
#define NGRAPH 64

__device__ __forceinline__ unsigned short f2bf(float f) {
  unsigned u = __float_as_uint(f);
  u += 0x7fffu + ((u >> 16) & 1u);          // round-to-nearest-even
  return (unsigned short)(u >> 16);
}

// CDNA5 async memory->LDS copy (16B per lane), tracked by ASYNCcnt.
__device__ __forceinline__ void async_ld_b128(unsigned lds_off, const void* gptr) {
  asm volatile("global_load_async_to_lds_b128 %0, %1, off"
               : : "v"(lds_off), "v"(gptr) : "memory");
}
__device__ __forceinline__ void s_wait_async0() {
  asm volatile("s_wait_asynccnt 0x0" ::: "memory");
}
__device__ __forceinline__ void s_wait_async6() {
  asm volatile("s_wait_asynccnt 0x6" ::: "memory");
}

// ---------------------------------------------------------------------------
// Generic bf16 x bf16 -> f32 GEMM with fused bias + activation.
// A:  M x K bf16 row-major.
// Wf: weights pre-laid-out in WMMA-B *fragment order*:
//       u32[K/32][N/16][32 lanes][8]
//     where word j of lane (half*16+l16) = {W[2*kg][n], W[2*kg+1][n]<<16},
//     kg = ks*16 + ((j&4)<<1) + half*4 + (j&3), n = ng*16 + l16.
//     => B tile for one k-step and 64 columns is a contiguous 4KB block, and
//        each fragment is two contiguous ds_load_b128 per lane (zero moves).
// Tile: 128(M) x 64(N) x 32(K) per 128-thread block, double-buffered LDS via
// global_load_async_to_lds_b128; each of 4 waves owns a 32-row stripe and
// issues 8 back-to-back v_wmma_f32_16x16x32_bf16 per K-step.
// act: 0=none, 1=relu, 2=elu.
// ---------------------------------------------------------------------------
__global__ __launch_bounds__(128) void gemm_bf16_kernel(
    const unsigned short* __restrict__ A,
    const unsigned*       __restrict__ Wf,
    const float*          __restrict__ bias,
    float*          __restrict__ Cf,
    unsigned short* __restrict__ Cb,
    int M, int K, int Nn, int act)
{
  // A tile: 128 rows x 32 bf16, row stride 40 shorts (80B, 16B aligned).
  __shared__ unsigned short lA[2][128][40];
  // B tile in fragment order: 4 n-groups x 32 lanes x 8 u32 = 4KB per buffer.
  __shared__ unsigned lBf[2][1024];

  const int tid  = threadIdx.x;
  const int wave = tid >> 5;
  const int lane = tid & 31;
  const int half = lane >> 4;
  const int l16  = lane & 15;
  const int m0   = blockIdx.x * 128;
  const int n0   = blockIdx.y * 64;
  const int ngrp = Nn >> 4;

  v8f acc[2][4];
  #pragma unroll
  for (int f = 0; f < 2; ++f)
    #pragma unroll
    for (int t = 0; t < 4; ++t) acc[f][t] = (v8f)(0.0f);

  // Async fill of one K-step's tiles (6 x 16B chunks per thread).
  auto issue = [&](int buf, int ks) {
    int k0 = ks << 5;
    #pragma unroll
    for (int j = 0; j < 4; ++j) {                  // A: 512 chunks of 16B
      int t  = tid + j * 128;
      int r  = t >> 2;
      int c8 = (t & 3) * 8;                        // bf16 column
      int gm = m0 + r; if (gm > M - 1) gm = M - 1; // clamp: row only feeds row
      async_ld_b128((unsigned)(size_t)&lA[buf][r][c8],
                    A + (size_t)gm * K + k0 + c8);
    }
    // B: one contiguous 4KB block -> 256 chunks of 16B, linear copy.
    const unsigned* gb = Wf + (((size_t)ks * ngrp + (n0 >> 4)) << 8);
    #pragma unroll
    for (int j = 0; j < 2; ++j) {
      int c = tid + j * 128;
      async_ld_b128((unsigned)(size_t)&lBf[buf][c * 4], gb + c * 4);
    }
  };

  const int nsteps = K >> 5;
  issue(0, 0);
  for (int i = 0; i < nsteps; ++i) {
    int cur = i & 1;
    if (i + 1 < nsteps) { issue(cur ^ 1, i + 1); s_wait_async6(); }
    else                { s_wait_async0(); }
    __syncthreads();

    // ---- preload all fragments: everything is b128 LDS traffic ----
    union { v16bf v; uint4 q[2]; } fa[2];
    #pragma unroll
    for (int f = 0; f < 2; ++f) {
      const unsigned short* rp = &lA[cur][wave * 32 + f * 16 + l16][0];
      fa[f].q[0] = *(const uint4*)(rp + half * 8);        // K = h*8 .. h*8+7
      fa[f].q[1] = *(const uint4*)(rp + 16 + half * 8);   // K = 16+h*8 ..
    }
    union { v16bf v; uint4 q[2]; } fb[4];
    #pragma unroll
    for (int t = 0; t < 4; ++t) {
      const unsigned* p = &lBf[cur][(t * 32 + lane) * 8];
      fb[t].q[0] = *(const uint4*)(p);
      fb[t].q[1] = *(const uint4*)(p + 4);
    }
    // ---- 8 back-to-back WMMAs ----
    #pragma unroll
    for (int f = 0; f < 2; ++f)
      #pragma unroll
      for (int t = 0; t < 4; ++t)
        acc[f][t] = __builtin_amdgcn_wmma_f32_16x16x32_bf16(
            false, fa[f].v, false, fb[t].v, (short)0, acc[f][t], false, false);
    __syncthreads();
  }

  // Epilogue: C/D layout -> VGPR r: lanes 0-15 M=r, lanes 16-31 M=r+8; N=l16.
  #pragma unroll
  for (int t = 0; t < 4; ++t) {
    int gn = n0 + t * 16 + l16;
    float bv = bias[gn];
    #pragma unroll
    for (int f = 0; f < 2; ++f) {
      #pragma unroll
      for (int r = 0; r < 8; ++r) {
        int gm = m0 + wave * 32 + f * 16 + half * 8 + r;
        if (gm < M) {
          float v = acc[f][t][r] + bv;
          if (act == 1)      v = fmaxf(v, 0.0f);
          else if (act == 2) v = (v > 0.0f) ? v : (__expf(v) - 1.0f);
          size_t o = (size_t)gm * Nn + gn;
          if (Cf) Cf[o] = v;
          if (Cb) Cb[o] = f2bf(v);
        }
      }
    }
  }
}

// --------------------------- helper kernels --------------------------------
// Convert f32 weights K x N into WMMA-B fragment-ordered layout (see above).
__global__ void cvt_w_frag_kernel(const float* __restrict__ src,
                                  unsigned* __restrict__ dst, int K, int N) {
  long total = (long)(K >> 1) * N;
  int ngrp = N >> 4;
  for (long t = blockIdx.x * (long)blockDim.x + threadIdx.x; t < total;
       t += (long)gridDim.x * blockDim.x) {
    int ks   = (int)(t >> 8) / ngrp;          // t / (ngrp*256)
    int rem  = (int)(t - ((long)ks * ngrp << 8));
    int ng   = rem >> 8;
    int lane = (rem >> 3) & 31;
    int j    = rem & 7;
    int half = lane >> 4, l16 = lane & 15;
    int kg   = ks * 16 + ((j & 4) << 1) + half * 4 + (j & 3);
    int n    = ng * 16 + l16;
    unsigned lo = f2bf(src[(size_t)(2 * kg) * N + n]);
    unsigned hi = f2bf(src[(size_t)(2 * kg + 1) * N + n]);
    dst[t] = lo | (hi << 16);
  }
}

__global__ void zero_kernel(float* p, long n) {
  for (long t = blockIdx.x * (long)blockDim.x + threadIdx.x; t < n;
       t += (long)gridDim.x * blockDim.x)
    p[t] = 0.0f;
}

__global__ void bn_stats_kernel(const float* __restrict__ src, int rows,
                                int cols, float* mean, float* rstd) {
  __shared__ float ssum[256], ssq[256];
  int c = blockIdx.x;
  float s = 0.f, q = 0.f;
  for (int r = threadIdx.x; r < rows; r += blockDim.x) {
    float v = src[(size_t)r * cols + c];
    s += v; q += v * v;
  }
  ssum[threadIdx.x] = s; ssq[threadIdx.x] = q;
  __syncthreads();
  for (int st = 128; st > 0; st >>= 1) {
    if (threadIdx.x < st) {
      ssum[threadIdx.x] += ssum[threadIdx.x + st];
      ssq[threadIdx.x]  += ssq[threadIdx.x + st];
    }
    __syncthreads();
  }
  if (threadIdx.x == 0) {
    float m = ssum[0] / rows;
    mean[c] = m;
    rstd[c] = rsqrtf(ssq[0] / rows - m * m + 1e-5f);
  }
}

__global__ void bn_apply_kernel(const float* __restrict__ src,
                                float* __restrict__ dst,
                                const float* mean, const float* rstd,
                                const float* g, const float* b,
                                long total, int cols) {
  for (long t = blockIdx.x * (long)blockDim.x + threadIdx.x; t < total;
       t += (long)gridDim.x * blockDim.x) {
    int c = (int)(t % cols);
    dst[t] = (src[t] - mean[c]) * rstd[c] * g[c] + b[c];
  }
}

// Gather + concat up to 3 f32 sources (optional row index per source) -> bf16.
__global__ void concat_kernel(unsigned short* __restrict__ dst, int M, int wt,
                              const float* s0, const int* i0, int w0,
                              const float* s1, const int* i1, int w1,
                              const float* s2, const int* i2, int w2) {
  long total = (long)M * wt;
  for (long t = blockIdx.x * (long)blockDim.x + threadIdx.x; t < total;
       t += (long)gridDim.x * blockDim.x) {
    int r = (int)(t / wt), c = (int)(t % wt);
    float v;
    if (c < w0)           { int rr = i0 ? i0[r] : r; v = s0[(size_t)rr * w0 + c]; }
    else if (c < w0 + w1) { int cc = c - w0; int rr = i1 ? i1[r] : r; v = s1[(size_t)rr * w1 + cc]; }
    else                  { int cc = c - w0 - w1; int rr = i2 ? i2[r] : r; v = s2[(size_t)rr * w2 + cc]; }
    dst[t] = f2bf(v);
  }
}

__global__ void scatter_add_kernel(const float* __restrict__ src,
                                   const int* __restrict__ idx,
                                   float* __restrict__ dst,
                                   float* __restrict__ cnt,
                                   long rows, int cols) {
  long total = rows * cols;
  for (long t = blockIdx.x * (long)blockDim.x + threadIdx.x; t < total;
       t += (long)gridDim.x * blockDim.x) {
    int r = (int)(t / cols), c = (int)(t % cols);
    int d = idx[r];
    atomicAdd(&dst[(size_t)d * cols + c], src[t]);
    if (c == 0) atomicAdd(&cnt[d], 1.0f);
  }
}

__global__ void seg_div_kernel(float* __restrict__ dst,
                               const float* __restrict__ cnt,
                               long total, int cols) {
  for (long t = blockIdx.x * (long)blockDim.x + threadIdx.x; t < total;
       t += (long)gridDim.x * blockDim.x) {
    int r = (int)(t / cols);
    dst[t] /= fmaxf(cnt[r], 1.0f);
  }
}

// ---------------------------------------------------------------------------
extern "C" void kernel_launch(void* const* d_in, const int* in_sizes, int n_in,
                              void* d_out, int out_size, void* d_ws, size_t ws_size,
                              hipStream_t stream) {
  (void)in_sizes; (void)out_size;
  if (n_in < 108) return;

  static const int EOd[6]  = {512, 512, 512, 512, 512, 128};
  static const int GId[6]  = {128, 256, 256, 256, 256, 256};
  static const int GOd[6]  = {128, 128, 128, 128, 128, 256};
  static const int N1Id[6] = {640, 640, 640, 640, 640, 256};

  const float* x_in = (const float*)d_in[0];
  const float* e_in = (const float*)d_in[1];
  const int*   eidx = (const int*)d_in[2];
  const int*   bat  = (const int*)d_in[3];
  const int*   row  = eidx;
  const int*   col  = eidx + NEDGE;

  // ---- workspace carve-up ----
  unsigned char* wsb = (unsigned char*)d_ws;
  size_t off = 0;
  auto alloc = [&](size_t bytes) -> void* {
    void* p = wsb + off;
    off = (off + bytes + 255) & ~(size_t)255;
    return p;
  };
  float*          x_f  = (float*)alloc((size_t)NNODE * 128 * 4);
  float*          e_f  = (float*)alloc((size_t)NEDGE * 512 * 4);
  unsigned short* cc   = (unsigned short*)alloc((size_t)NEDGE * 768 * 2);
  unsigned short* hid  = (unsigned short*)alloc((size_t)NEDGE * 512 * 2);
  float*          m_f  = (float*)cc;   // alias: messages live where concat was
  float*          agg  = (float*)alloc((size_t)NNODE * 128 * 4);
  float*          cnt  = (float*)alloc((size_t)NNODE * 4);
  float*          xg   = (float*)alloc((size_t)NGRAPH * 128 * 4);
  float*          cg   = (float*)alloc((size_t)NGRAPH * 4);
  float*          u_f  = (float*)alloc((size_t)NGRAPH * 256 * 4);
  unsigned short* u_b  = (unsigned short*)alloc((size_t)NGRAPH * 256 * 2);
  float*          stat = (float*)alloc(512 * 4);
  unsigned*       wbuf = (unsigned*)alloc((size_t)12 * 1024 * 1024);
  if (ws_size < off) return;

  // ---- launch helpers ----
  auto gemm = [&](const unsigned short* A, const unsigned* Wf,
                  const float* bias, float* Cf, unsigned short* Cb,
                  int M, int K, int Nw, int act) {
    dim3 grid((M + 127) / 128, Nw / 64);
    gemm_bf16_kernel<<<grid, 128, 0, stream>>>(A, Wf, bias, Cf, Cb, M, K, Nw, act);
  };
  auto zero = [&](float* p, long n) {
    int b = (int)((n + 255) / 256); if (b > 4096) b = 4096;
    zero_kernel<<<b, 256, 0, stream>>>(p, n);
  };
  auto concat = [&](unsigned short* dst, int M, int wt,
                    const float* s0, const int* i0, int w0,
                    const float* s1, const int* i1, int w1,
                    const float* s2, const int* i2, int w2) {
    long tot = (long)M * wt;
    int b = (int)((tot + 255) / 256); if (b > 8192) b = 8192;
    concat_kernel<<<b, 256, 0, stream>>>(dst, M, wt, s0, i0, w0, s1, i1, w1, s2, i2, w2);
  };
  auto scat = [&](const float* src, const int* idx, float* dst, float* c,
                  long rows, int cols) {
    long tot = rows * cols;
    int b = (int)((tot + 255) / 256); if (b > 8192) b = 8192;
    scatter_add_kernel<<<b, 256, 0, stream>>>(src, idx, dst, c, rows, cols);
  };
  auto segdiv = [&](float* dst, const float* c, long rows, int cols) {
    long tot = rows * cols;
    int b = (int)((tot + 255) / 256); if (b > 8192) b = 8192;
    seg_div_kernel<<<b, 256, 0, stream>>>(dst, c, tot, cols);
  };

  // ---- convert all weights to fragment-ordered bf16 once ----
  size_t woff = 0;
  auto cvtW = [&](int idx, int k, int n) -> unsigned* {
    unsigned* dst = wbuf + woff;
    long tot = (long)(k >> 1) * n;
    woff = (woff + (size_t)tot + 3) & ~(size_t)3;   // keep 16B alignment
    int b = (int)((tot + 255) / 256); if (b > 2048) b = 2048;
    cvt_w_frag_kernel<<<b, 256, 0, stream>>>((const float*)d_in[idx], dst, k, n);
    return dst;
  };

  unsigned *We1[6], *We2[6], *Wn11[6], *Wn12[6], *Wn21[6], *Wn22[6], *Wg1[6], *Wg2[6];
  const float *Be1[6], *Be2[6], *Bn11[6], *Bn12[6], *Bn21[6], *Bn22[6], *Bg1[6], *Bg2[6];
  for (int i = 0; i < 6; ++i) {
    int base = 8 + i * 16;
    int ein  = (i == 0) ? 320 : 768;
    We1[i]  = cvtW(base + 0,  ein,     EOd[i]); Be1[i]  = (const float*)d_in[base + 1];
    We2[i]  = cvtW(base + 2,  EOd[i],  EOd[i]); Be2[i]  = (const float*)d_in[base + 3];
    Wn11[i] = cvtW(base + 4,  N1Id[i], 128);    Bn11[i] = (const float*)d_in[base + 5];
    Wn12[i] = cvtW(base + 6,  128,     128);    Bn12[i] = (const float*)d_in[base + 7];
    Wn21[i] = cvtW(base + 8,  256,     128);    Bn21[i] = (const float*)d_in[base + 9];
    Wn22[i] = cvtW(base + 10, 128,     128);    Bn22[i] = (const float*)d_in[base + 11];
    Wg1[i]  = cvtW(base + 12, GId[i],  GOd[i]); Bg1[i]  = (const float*)d_in[base + 13];
    Wg2[i]  = cvtW(base + 14, GOd[i],  GOd[i]); Bg2[i]  = (const float*)d_in[base + 15];
  }
  unsigned* Wl1 = cvtW(104, 256, 512); const float* Bl1 = (const float*)d_in[105];
  unsigned* Wl2 = cvtW(106, 512, 128); const float* Bl2 = (const float*)d_in[107];

  // ---- batch norm on x and edge_attr ----
  float* mean_x = stat;       float* rstd_x = stat + 128;
  float* mean_e = stat + 256; float* rstd_e = stat + 320;
  bn_stats_kernel<<<128, 256, 0, stream>>>(x_in, NNODE, 128, mean_x, rstd_x);
  bn_apply_kernel<<<4096, 256, 0, stream>>>(x_in, x_f, mean_x, rstd_x,
      (const float*)d_in[4], (const float*)d_in[5], (long)NNODE * 128, 128);
  bn_stats_kernel<<<64, 256, 0, stream>>>(e_in, NEDGE, 64, mean_e, rstd_e);
  bn_apply_kernel<<<4096, 256, 0, stream>>>(e_in, e_f, mean_e, rstd_e,
      (const float*)d_in[6], (const float*)d_in[7], (long)NEDGE * 64, 64);

  // ---- MetaLayers ----
  int ew = 64;
  for (int i = 0; i < 6; ++i) {
    int eo = EOd[i], gi = GId[i], go = GOd[i];
    // EdgeModel: e = MLP([x[row], x[col], e])
    concat(cc, NEDGE, 256 + ew, x_f, row, 128, x_f, col, 128, e_f, nullptr, ew);
    gemm(cc,  We1[i], Be1[i], nullptr, hid, NEDGE, 256 + ew, eo, 1);
    gemm(hid, We2[i], Be2[i], e_f, nullptr, NEDGE, eo, eo, 0);
    ew = eo;
    // NodeModel part 1: m = MLP([x[row], e]); agg = scatter_mean(m, col)
    concat(cc, NEDGE, 128 + ew, x_f, row, 128, e_f, nullptr, ew, nullptr, nullptr, 0);
    gemm(cc,  Wn11[i], Bn11[i], nullptr, hid, NEDGE, 128 + ew, 128, 1);
    gemm(hid, Wn12[i], Bn12[i], m_f, nullptr, NEDGE, 128, 128, 0);
    zero(agg, (long)NNODE * 128); zero(cnt, NNODE);
    scat(m_f, col, agg, cnt, NEDGE, 128);
    segdiv(agg, cnt, NNODE, 128);
    // NodeModel part 2: x = MLP([x, agg])
    concat(cc, NNODE, 256, x_f, nullptr, 128, agg, nullptr, 128, nullptr, nullptr, 0);
    gemm(cc,  Wn21[i], Bn21[i], nullptr, hid, NNODE, 256, 128, 1);
    gemm(hid, Wn22[i], Bn22[i], x_f, nullptr, NNODE, 128, 128, 0);
    // GlobalModel: u = MLP([u, seg_mean(x, batch)])
    zero(xg, (long)NGRAPH * 128); zero(cg, NGRAPH);
    scat(x_f, bat, xg, cg, NNODE, 128);
    segdiv(xg, cg, NGRAPH, 128);
    if (i == 0)
      concat(cc, NGRAPH, 128, xg, nullptr, 128, nullptr, nullptr, 0, nullptr, nullptr, 0);
    else
      concat(cc, NGRAPH, gi, u_f, nullptr, gi - 128, xg, nullptr, 128, nullptr, nullptr, 0);
    gemm(cc, Wg1[i], Bg1[i], nullptr, hid, NGRAPH, gi, go, 1);
    bool last = (i == 5);
    gemm(hid, Wg2[i], Bg2[i], last ? (float*)d_out : u_f, last ? u_b : nullptr,
         NGRAPH, go, go, 0);
  }

  // ---- head: y = elu(u @ lin1 + b1) @ lin2 + b2 ----
  gemm(u_b, Wl1, Bl1, nullptr, hid, NGRAPH, 256, 512, 2);
  gemm(hid, Wl2, Bl2, (float*)d_out + NGRAPH * 256, nullptr, NGRAPH, 512, 128, 0);
}
